// CosineContrastiveLoss_NoExp_77738908057914
// MI455X (gfx1250) — matrix operand order repaired and checked by reading, not compiled, not run on hardware
//
#include <hip/hip_runtime.h>
#include <hip/hip_bf16.h>
#include <math.h>

// ---------------------------------------------------------------------------
// Problem: B=32, C=1, H=512, W=512.  HW = 262144 per sample.
// Kernel 1: streaming elementwise (sigmoid) + 5 per-sample reductions.
// Kernel 2: 32x32x262144 Gram matrix  G[b,d] = sum_k sq1[b,k]*sq2[d,k]
//           via v_wmma_f32_16x16x32_bf16 (2x2 tiles of 16x16, K tiled by 32).
// Kernel 3: 1-wave finalize -> scalar loss.
//
// Memory-bound: 96 MB of fp32 inputs @ 23.3 TB/s ~= 4.1 us floor. The Gram
// re-read of input1/input2 (64 MB) hits the 192 MB L2. Sigmoid uses
// hardware v_rcp_f32 (1 ulp) instead of the precise-divide sequence so the
// Gram kernel stays memory-bound rather than VALU-bound (results are
// truncated to bf16 anyway).
// ---------------------------------------------------------------------------

typedef __attribute__((ext_vector_type(16))) __bf16 v16bf;
typedef __attribute__((ext_vector_type(8)))  float  v8f;

#define HW 262144
#define NB 32

__device__ __forceinline__ float sigm(float x) {
    // 1 / (1 + exp(-x)) with hardware rcp (v_rcp_f32, ~1 ulp)
    return __builtin_amdgcn_rcpf(1.0f + __expf(-x));
}

// ---------------------------------------------------------------------------
// Kernel 1: per-sample reductions.
// grid = (64 chunks, 32 samples), block = 256 threads (8 waves).
// Each block covers 4096 contiguous elements of one sample (float4 loads).
// sums layout in ws (floats): [0..31]=pp2 [32..63]=p1 [64..95]=p2
//                             [96..127]=q1 [128..159]=q2
// ---------------------------------------------------------------------------
__global__ void ccl_sums_kernel(const float* __restrict__ in1,
                                const float* __restrict__ in2,
                                const float* __restrict__ msk,
                                float* __restrict__ sums) {
    const int b     = blockIdx.y;
    const size_t base = (size_t)b * HW + (size_t)blockIdx.x * 4096;
    const int t = threadIdx.x;

    float app = 0.f, ap1 = 0.f, ap2 = 0.f, aq1 = 0.f, aq2 = 0.f;
#pragma unroll
    for (int i = 0; i < 4; ++i) {
        size_t off = base + (size_t)i * 1024 + (size_t)t * 4;
        float4 x1 = *(const float4*)(in1 + off);
        float4 x2 = *(const float4*)(in2 + off);
        float4 mm = *(const float4*)(msk + off);
        float xs1[4] = {x1.x, x1.y, x1.z, x1.w};
        float xs2[4] = {x2.x, x2.y, x2.z, x2.w};
        float ms[4]  = {mm.x, mm.y, mm.z, mm.w};
#pragma unroll
        for (int j = 0; j < 4; ++j) {
            float s1 = sigm(xs1[j]);
            float s2 = sigm(xs2[j]);
            float m  = ms[j];
            float pp = m * s1 * s2;
            float p1 = m * s1;
            float p2 = m * s2;
            app += pp * pp;
            ap1 += p1 * p1;
            ap2 += p2 * p2;
            aq1 += s1 * s1;
            aq2 += s2 * s2;
        }
    }
    // wave32 tree reduction
#pragma unroll
    for (int o = 16; o > 0; o >>= 1) {
        app += __shfl_down(app, o);
        ap1 += __shfl_down(ap1, o);
        ap2 += __shfl_down(ap2, o);
        aq1 += __shfl_down(aq1, o);
        aq2 += __shfl_down(aq2, o);
    }
    if ((t & 31) == 0) {
        atomicAdd(&sums[      b], app);
        atomicAdd(&sums[ 32 + b], ap1);
        atomicAdd(&sums[ 64 + b], ap2);
        atomicAdd(&sums[ 96 + b], aq1);
        atomicAdd(&sums[128 + b], aq2);
    }
}

// ---------------------------------------------------------------------------
// Kernel 2: Gram matrix via WMMA bf16.
// grid = 128 blocks x 256 threads (8 waves).  Each wave owns 256 K values
// (8 WMMA steps of K=32) and accumulates the full 32x32 tile set:
//   acc[I][J]  with A_I = sq1 rows 16I..16I+15, B_J = sq2 rows 16J..16J+15.
// ---------------------------------------------------------------------------
__global__ void ccl_gram_kernel(const float* __restrict__ in1,
                                const float* __restrict__ in2,
                                float* __restrict__ G) {
    const int lane = threadIdx.x & 31;
    const int wave = threadIdx.x >> 5;
    const int half = lane >> 4;     // which 16-lane half
    const int r    = lane & 15;     // row within a 16-row tile
    const int kbase = (blockIdx.x * 8 + wave) * 256;

    v8f acc00 = {}, acc01 = {}, acc10 = {}, acc11 = {};

    for (int ks = 0; ks < 256; ks += 32) {
        const int k0 = kbase + ks;
        v16bf a0, a1, b0, b1;

        // A-matrix 16x32 bf16 layout:
        // lane half 0: elems 0..15 = K k0+0..15
        // lane half 1: elems 0..15 = K k0+16..31 (contiguous)
        {
            const float* p = in1 + (size_t)r * HW + k0 + half * 16;
#pragma unroll
            for (int i = 0; i < 8; ++i) { float s = sigm(p[i]);      a0[i]     = (__bf16)(s * s); }
#pragma unroll
            for (int i = 0; i < 8; ++i) { float s = sigm(p[8 + i]); a0[8 + i] = (__bf16)(s * s); }
        }
        {
            const float* p = in1 + (size_t)(16 + r) * HW + k0 + half * 16;
#pragma unroll
            for (int i = 0; i < 8; ++i) { float s = sigm(p[i]);      a1[i]     = (__bf16)(s * s); }
#pragma unroll
            for (int i = 0; i < 8; ++i) { float s = sigm(p[8 + i]); a1[8 + i] = (__bf16)(s * s); }
        }
        // B-matrix 32x16 bf16 layout: lane n holds column n (= sq2 row n);
        // lanes 0-15: K k0+0..15, lanes 16-31: K k0+16..31 (contiguous).
        {
            const float* p = in2 + (size_t)r * HW + k0 + half * 16;
#pragma unroll
            for (int i = 0; i < 16; ++i) { float s = sigm(p[i]); b0[i] = (__bf16)(s * s); }
        }
        {
            const float* p = in2 + (size_t)(16 + r) * HW + k0 + half * 16;
#pragma unroll
            for (int i = 0; i < 16; ++i) { float s = sigm(p[i]); b1[i] = (__bf16)(s * s); }
        }

        // D = A x B + C   (8 args: neg_a, A, neg_b, B, c_mod, C, reuse_a, reuse_b)
        acc00 = __builtin_amdgcn_wmma_f32_16x16x32_bf16(false, a0, false, b0, (short)0, acc00, false, false);
        acc01 = __builtin_amdgcn_wmma_f32_16x16x32_bf16(false, a0, false, b1, (short)0, acc01, false, false);
        acc10 = __builtin_amdgcn_wmma_f32_16x16x32_bf16(false, a1, false, b0, (short)0, acc10, false, false);
        acc11 = __builtin_amdgcn_wmma_f32_16x16x32_bf16(false, a1, false, b1, (short)0, acc11, false, false);
    }

    // C/D layout: VGPR rr, lane l -> row rr + 8*half, col l&15
    const int d0 = lane & 15;
#pragma unroll
    for (int rr = 0; rr < 8; ++rr) {
        const int brow = rr + 8 * half;
        atomicAdd(&G[(brow)      * NB +      d0], acc00[rr]);
        atomicAdd(&G[(brow)      * NB + 16 + d0], acc01[rr]);
        atomicAdd(&G[(16 + brow) * NB +      d0], acc10[rr]);
        atomicAdd(&G[(16 + brow) * NB + 16 + d0], acc11[rr]);
    }
}

// ---------------------------------------------------------------------------
// Kernel 3: finalize. 1 block, 32 threads (one wave), lane = sample b.
// sim_pos[b] = sqrt(pp2)/(sqrt(p1)*sqrt(p2))
// sim_neg[b] = sum_{d!=b} sqrt(G[b,d]) / (sqrt(q1[b])*sqrt(q2[d]))
// loss = mean_{i,j} -log(sp[j]/(sp[j]+sn[i]))
// ---------------------------------------------------------------------------
__global__ void ccl_final_kernel(const float* __restrict__ sums,
                                 const float* __restrict__ G,
                                 float* __restrict__ out) {
    __shared__ float n2s[NB], sps[NB], sns[NB], red[NB];
    const int b = threadIdx.x;

    float pp2 = sums[b];
    float p1  = sums[32 + b];
    float p2  = sums[64 + b];
    float q1  = sums[96 + b];
    float q2  = sums[128 + b];

    float sp = sqrtf(pp2) / (sqrtf(p1) * sqrtf(p2));
    float n1 = sqrtf(q1);
    n2s[b] = sqrtf(q2);
    sps[b] = sp;
    __syncthreads();

    float sn = 0.f;
    for (int d = 0; d < NB; ++d)
        if (d != b) sn += sqrtf(G[b * NB + d]) / (n1 * n2s[d]);
    sns[b] = sn;
    __syncthreads();

    // lane j accumulates sum_i -log(sp[j]/(sp[j]+sn[i]))
    float spj = sps[b];
    float part = 0.f;
    for (int i = 0; i < NB; ++i)
        part += -logf(spj / (spj + sns[i]));
    red[b] = part;
    __syncthreads();

    if (b == 0) {
        float t = 0.f;
        for (int i = 0; i < NB; ++i) t += red[i];
        out[0] = t / 1024.0f;
    }
}

// ---------------------------------------------------------------------------
extern "C" void kernel_launch(void* const* d_in, const int* in_sizes, int n_in,
                              void* d_out, int out_size, void* d_ws, size_t ws_size,
                              hipStream_t stream) {
    const float* in1 = (const float*)d_in[0];
    const float* in2 = (const float*)d_in[1];
    const float* msk = (const float*)d_in[2];
    float* out = (float*)d_out;

    float* sums = (float*)d_ws;          // 160 floats
    float* G    = sums + 160;            // 1024 floats

    hipMemsetAsync(d_ws, 0, (160 + NB * NB) * sizeof(float), stream);

    dim3 gridA(64, NB);                  // 64 chunks x 32 samples
    ccl_sums_kernel<<<gridA, 256, 0, stream>>>(in1, in2, msk, sums);

    ccl_gram_kernel<<<128, 256, 0, stream>>>(in1, in2, G);

    ccl_final_kernel<<<1, NB, 0, stream>>>(sums, G, out);
}
